// CrossSectionalRanker_14362370637898
// MI455X (gfx1250) — matrix operand order repaired and checked by reading, not compiled, not run on hardware
//
#include <hip/hip_runtime.h>
#include <hip/hip_bf16.h>
#include <math.h>

// ---------------------------------------------------------------------------
// CrossSectionalRanker for MI455X (gfx1250, wave32, WMMA).
// Dense tower (~1.5 TFLOP of GEMM over 131072 rows) runs on
// v_wmma_f32_16x16x32_bf16, register-blocked 2x4 tiles per wave (8 WMMAs per
// 12 b128 fragment loads) with fused epilogues. The 512-wide LayerNorm+GELU
// stage transitions fuse the row reduction through LDS. Shortlist rerank
// (<1% of FLOPs) runs as simple f32 VALU kernels + LDS bitonic top-k.
// ---------------------------------------------------------------------------

#define D_DAYS 64
#define GROUP  2048
#define NROWS  (D_DAYS * GROUP)   // 131072
#define INDIM  256
#define HDIM   512
#define RDIM   128
#define KSHORT 128
#define TOKENS (D_DAYS * KSHORT)  // 8192
#define RKPAD  544                // 514 padded to a multiple of 32
#define LN_EPS 1e-5f

typedef unsigned short u16;
typedef unsigned int   u32;
typedef __attribute__((ext_vector_type(16))) __bf16 bf16x16;
typedef __attribute__((ext_vector_type(8)))  float  f32x8;

union FragB { bf16x16 v; uint4 q[2]; };

__device__ __forceinline__ u16 f2bf(float f) {
  u32 u = __float_as_uint(f);
  u32 r = ((u >> 16) & 1u) + 0x7fffu;   // round-to-nearest-even
  return (u16)((u + r) >> 16);
}
__device__ __forceinline__ float bf2f(u16 h) { return __uint_as_float(((u32)h) << 16); }
__device__ __forceinline__ float gelu_f(float x) { return 0.5f * x * (1.f + erff(x * 0.70710678118654752f)); }
__device__ __forceinline__ float sigmoid_f(float x) { return 1.f / (1.f + expf(-x)); }

__device__ __forceinline__ float block_reduce_256(float v, float* s) {
  int t = threadIdx.x;
  s[t] = v; __syncthreads();
#pragma unroll
  for (int o = 128; o > 0; o >>= 1) { if (t < o) s[t] += s[t + o]; __syncthreads(); }
  float r = s[0]; __syncthreads();
  return r;
}

// ---- weight packing: f32 [K, NOUT] row-major -> bf16 B^T [NOUT, KPAD] ------
__global__ __launch_bounds__(256) void pack_w_kernel(const float* __restrict__ src,
    u16* __restrict__ dst, int K, int NOUT, int KPAD) {
  size_t i = (size_t)blockIdx.x * 256 + threadIdx.x;
  size_t total = (size_t)NOUT * KPAD;
  if (i >= total) return;
  int n = (int)(i / KPAD), k = (int)(i % KPAD);
  dst[i] = (k < K) ? f2bf(src[(size_t)k * NOUT + n]) : (u16)0;
}

// ---- encode: LN(features) -> xn bf16, linear_score ------------------------
__global__ __launch_bounds__(256) void encode_kernel(const float* __restrict__ feat,
    const float* __restrict__ g, const float* __restrict__ b,
    const float* __restrict__ lw, u16* __restrict__ xn, float* __restrict__ lin) {
  __shared__ float s[256];
  int row = blockIdx.x, t = threadIdx.x;
  size_t base = (size_t)row * INDIM;
  float v = feat[base + t];
  float m = block_reduce_256(v, s) * (1.f / INDIM);
  float d = v - m;
  float var = block_reduce_256(d * d, s) * (1.f / INDIM);
  float xv = d * rsqrtf(var + LN_EPS) * g[t] + b[t];
  xn[base + t] = f2bf(xv);
  float dot = block_reduce_256(xv * lw[t], s);
  if (t == 0) lin[row] = dot;
}

// ---- generic row LayerNorm (width W), optional bf16 and/or f32 outputs ----
__global__ __launch_bounds__(256) void ln_rows_kernel(const float* __restrict__ x,
    const float* __restrict__ g, const float* __restrict__ b,
    u16* __restrict__ obf, float* __restrict__ of, int W) {
  __shared__ float s[256];
  int row = blockIdx.x, t = threadIdx.x;
  size_t base = (size_t)row * W;
  float sum = 0.f;
  for (int c = t; c < W; c += 256) sum += x[base + c];
  float m = block_reduce_256(sum, s) / W;
  float sq = 0.f;
  for (int c = t; c < W; c += 256) { float d = x[base + c] - m; sq += d * d; }
  float rs = rsqrtf(block_reduce_256(sq, s) / W + LN_EPS);
  for (int c = t; c < W; c += 256) {
    float v = (x[base + c] - m) * rs * g[c] + b[c];
    if (obf) obf[base + c] = f2bf(v);
    if (of)  of[base + c]  = v;
  }
}

// ---- LN(latent) + broad head ----------------------------------------------
__global__ __launch_bounds__(256) void latent_broad_kernel(const float* __restrict__ x,
    const float* __restrict__ g, const float* __restrict__ b,
    const float* __restrict__ bw, const float* __restrict__ bb,
    const float* __restrict__ lsp, const float* __restrict__ lin,
    u16* __restrict__ latent, float* __restrict__ broad) {
  __shared__ float s[256];
  int row = blockIdx.x, t = threadIdx.x;
  size_t base = (size_t)row * HDIM;
  float sum = 0.f;
  for (int c = t; c < HDIM; c += 256) sum += x[base + c];
  float m = block_reduce_256(sum, s) * (1.f / HDIM);
  float sq = 0.f;
  for (int c = t; c < HDIM; c += 256) { float d = x[base + c] - m; sq += d * d; }
  float rs = rsqrtf(block_reduce_256(sq, s) * (1.f / HDIM) + LN_EPS);
  float dot = 0.f;
  for (int c = t; c < HDIM; c += 256) {
    float v = (x[base + c] - m) * rs * g[c] + b[c];
    latent[base + c] = f2bf(v);
    dot += v * bw[c];
  }
  float tot = block_reduce_256(dot, s);
  if (t == 0) {
    float ls = fminf(fmaxf(lsp[0], 0.f), 2.f);
    broad[row] = tot + bb[0] + ls * lin[row];
  }
}

// ---------------------------------------------------------------------------
// Register-blocked WMMA GEMM: workgroup owns 32 rows; each wave owns a
// 16*NB-wide column strip and computes a 2xNB tile block: one k-step =
// 2 A fragments + NB B fragments -> 2*NB WMMAs (A/B fragments reused).
// EPI: 0 = store f32; 1 = GELU -> bf16; 2 = x*(0.75+0.5*sigmoid(.)) -> bf16;
//      3 = residual gate: ef0 + sigmoid(.)*ef1 -> f32+bf16
// ---------------------------------------------------------------------------
template<int EPI, int NB>
__global__ __launch_bounds__(256) void wmma_mb32_kernel(
    const u16* __restrict__ A, int K,
    const u16* __restrict__ WT, const float* __restrict__ bias, int NOUT,
    float* __restrict__ outF, u16* __restrict__ outB,
    const u16* __restrict__ ebf, const float* __restrict__ ef0,
    const float* __restrict__ ef1) {
  const int tid = threadIdx.x, wave = tid >> 5, lane = tid & 31;
  const int m0 = blockIdx.x * 32;
  const int lrowoff = (lane & 16) ? 8 : 0;
  // A fragments: lane holds row; K chunks [kb+kbase,+8) and [kb+kbase+16,+8)
  const u16* aptr0 = A + (size_t)(m0 + (lane & 15)) * K + ((lane >> 4) << 3);
  const u16* aptr1 = aptr0 + (size_t)16 * K;

  for (int g = wave; g * (16 * NB) < NOUT; g += 8) {
    const int n0 = g * (16 * NB);
    const u16* wptr[NB];
#pragma unroll
    for (int j = 0; j < NB; j++)
      wptr[j] = WT + (size_t)(n0 + 16 * j + (lane & 15)) * K + ((lane >> 4) << 3);

    f32x8 acc[2 * NB];
#pragma unroll
    for (int tt = 0; tt < 2 * NB; tt++)
#pragma unroll
      for (int i = 0; i < 8; i++) acc[tt][i] = 0.f;

    for (int kb = 0; kb < K; kb += 32) {
      FragB fa0, fa1, fb[NB];
      fa0.q[0] = *(const uint4*)(aptr0 + kb);
      fa0.q[1] = *(const uint4*)(aptr0 + kb + 16);
      fa1.q[0] = *(const uint4*)(aptr1 + kb);
      fa1.q[1] = *(const uint4*)(aptr1 + kb + 16);
#pragma unroll
      for (int j = 0; j < NB; j++) {
        fb[j].q[0] = *(const uint4*)(wptr[j] + kb);
        fb[j].q[1] = *(const uint4*)(wptr[j] + kb + 16);
      }
#pragma unroll
      for (int j = 0; j < NB; j++) {
        acc[2 * j]     = __builtin_amdgcn_wmma_f32_16x16x32_bf16(
            false, fa0.v, false, fb[j].v, (short)0, acc[2 * j], false, false);
        acc[2 * j + 1] = __builtin_amdgcn_wmma_f32_16x16x32_bf16(
            false, fa1.v, false, fb[j].v, (short)0, acc[2 * j + 1], false, false);
      }
    }

#pragma unroll
    for (int j = 0; j < NB; j++) {
      const int col = n0 + 16 * j + (lane & 15);
      const float bv = bias ? bias[col] : 0.f;
#pragma unroll
      for (int mi = 0; mi < 2; mi++) {
#pragma unroll
        for (int r = 0; r < 8; r++) {
          const int mrow = m0 + 16 * mi + r + lrowoff;
          const size_t o = (size_t)mrow * NOUT + col;
          float v = acc[2 * j + mi][r] + bv;
          if constexpr (EPI == 0) { outF[o] = v; }
          else if constexpr (EPI == 1) { outB[o] = f2bf(gelu_f(v)); }
          else if constexpr (EPI == 2) {
            float xv = bf2f(ebf[o]);
            outB[o] = f2bf(xv * (0.75f + 0.5f * sigmoid_f(v)));
          } else if constexpr (EPI == 3) {
            float nx = ef0[o] + sigmoid_f(v) * ef1[o];
            outF[o] = nx; outB[o] = f2bf(nx);
          }
        }
      }
    }
  }
}

// ---------------------------------------------------------------------------
// WMMA GEMM with fused row LayerNorm + GELU epilogue. Workgroup owns 16 rows
// x NT cols (NT=512): wave w computes a 1x4 tile block (cols w*64..w*64+63),
// one A fragment feeds 4 WMMAs per k-step; row stats reduced through LDS.
// ---------------------------------------------------------------------------
template<int NT>
__global__ __launch_bounds__(256) void wmma_lngelu_kernel(
    const u16* __restrict__ A, int K,
    const u16* __restrict__ WT, const float* __restrict__ bias,
    float* __restrict__ outF, u16* __restrict__ outB,
    const float* __restrict__ lng, const float* __restrict__ lnb) {
  __shared__ float smem[16 * NT];
  __shared__ float psum[256];
  __shared__ float psq[256];
  __shared__ float mean[16], rstd[16];

  const int tid = threadIdx.x, wave = tid >> 5, lane = tid & 31;
  const int m0 = blockIdx.x * 16;
  const int lrowoff = (lane & 16) ? 8 : 0;
  const u16* aptr = A + (size_t)(m0 + (lane & 15)) * K + ((lane >> 4) << 3);
  const int n0 = wave * (NT / 8);   // 64 columns per wave

  const u16* wptr[4];
#pragma unroll
  for (int j = 0; j < 4; j++)
    wptr[j] = WT + (size_t)(n0 + 16 * j + (lane & 15)) * K + ((lane >> 4) << 3);

  f32x8 acc[4];
#pragma unroll
  for (int tt = 0; tt < 4; tt++)
#pragma unroll
    for (int i = 0; i < 8; i++) acc[tt][i] = 0.f;

  for (int kb = 0; kb < K; kb += 32) {
    FragB fa, fb[4];
    fa.q[0] = *(const uint4*)(aptr + kb);
    fa.q[1] = *(const uint4*)(aptr + kb + 16);
#pragma unroll
    for (int j = 0; j < 4; j++) {
      fb[j].q[0] = *(const uint4*)(wptr[j] + kb);
      fb[j].q[1] = *(const uint4*)(wptr[j] + kb + 16);
    }
#pragma unroll
    for (int j = 0; j < 4; j++)
      acc[j] = __builtin_amdgcn_wmma_f32_16x16x32_bf16(
          false, fa.v, false, fb[j].v, (short)0, acc[j], false, false);
  }

#pragma unroll
  for (int j = 0; j < 4; j++) {
    const int col = n0 + 16 * j + (lane & 15);
    const float bv = bias ? bias[col] : 0.f;
#pragma unroll
    for (int r = 0; r < 8; r++) smem[(r + lrowoff) * NT + col] = acc[j][r] + bv;
  }
  __syncthreads();
  {
    int rr = tid >> 4, sub = tid & 15;
    float sum = 0.f, sq = 0.f;
    for (int cq = sub; cq < NT; cq += 16) { float v = smem[rr * NT + cq]; sum += v; sq += v * v; }
    psum[rr * 16 + sub] = sum; psq[rr * 16 + sub] = sq;
  }
  __syncthreads();
  if (tid < 16) {
    float s = 0.f, q = 0.f;
#pragma unroll
    for (int j = 0; j < 16; j++) { s += psum[tid * 16 + j]; q += psq[tid * 16 + j]; }
    float m = s / NT;
    mean[tid] = m;
    rstd[tid] = rsqrtf(q / NT - m * m + LN_EPS);
  }
  __syncthreads();
  for (int e = tid; e < 16 * NT; e += 256) {
    int r = e / NT, cc = e - r * NT;
    float v = (smem[e] - mean[r]) * rstd[r] * lng[cc] + lnb[cc];
    v = gelu_f(v);
    size_t o = (size_t)(m0 + r) * NT + cc;
    outF[o] = v; outB[o] = f2bf(v);
  }
}

// ---- per-day top-128 via LDS bitonic sort (descending) --------------------
__global__ __launch_bounds__(256) void topk_kernel(const float* __restrict__ broad,
                                                   int* __restrict__ idx) {
  __shared__ float v[GROUP];
  __shared__ int  ix[GROUP];
  int d = blockIdx.x, t = threadIdx.x;
  for (int i = t; i < GROUP; i += 256) { v[i] = broad[(size_t)d * GROUP + i]; ix[i] = i; }
  __syncthreads();
  for (int k = 2; k <= GROUP; k <<= 1) {
    for (int j = k >> 1; j > 0; j >>= 1) {
      for (int i = t; i < GROUP; i += 256) {
        int p = i ^ j;
        if (p > i) {
          bool desc = ((i & k) == 0);
          float a = v[i], bb = v[p];
          bool sw = desc ? (a < bb) : (a > bb);
          if (sw) { v[i] = bb; v[p] = a; int q = ix[i]; ix[i] = ix[p]; ix[p] = q; }
        }
      }
      __syncthreads();
    }
  }
  for (int i = t; i < KSHORT; i += 256) idx[d * KSHORT + i] = ix[i];
}

// ---- gather shortlist rows into K-padded bf16 rerank input ----------------
__global__ __launch_bounds__(256) void gather_pack_kernel(const u16* __restrict__ latent,
    const float* __restrict__ broad, const float* __restrict__ lin,
    const int* __restrict__ idx, u16* __restrict__ slin) {
  int tkn = blockIdx.x, day = tkn >> 7;
  int src = day * GROUP + idx[tkn];
  size_t sb = (size_t)src * HDIM, db = (size_t)tkn * RKPAD;
  for (int c = threadIdx.x; c < RKPAD; c += 256) {
    u16 v;
    if (c < HDIM)            v = latent[sb + c];
    else if (c == HDIM)      v = f2bf(broad[src]);
    else if (c == HDIM + 1)  v = f2bf(lin[src]);
    else                     v = 0;
    slin[db + c] = v;
  }
}

// ---- small f32 GEMM (shortlist path), W is reference [K,N] row-major ------
template<int ACT>  // 0 = none, 1 = gelu (before residual)
__global__ __launch_bounds__(256) void gemm_f32_kernel(const float* __restrict__ A,
    const float* __restrict__ W, const float* __restrict__ bias,
    const float* __restrict__ res, float* __restrict__ out, int M, int K, int N) {
  size_t i = (size_t)blockIdx.x * 256 + threadIdx.x;
  if (i >= (size_t)M * N) return;
  int row = (int)(i / N), col = (int)(i % N);
  const float* a = A + (size_t)row * K;
  float acc = bias ? bias[col] : 0.f;
  for (int k = 0; k < K; k++) acc += a[k] * W[(size_t)k * N + col];
  if constexpr (ACT == 1) acc = gelu_f(acc);
  if (res) acc += res[i];
  out[i] = acc;
}

// ---- shortlist MHA: one block per (day, head), K/V staged in LDS ----------
__global__ __launch_bounds__(128) void attn_kernel(const float* __restrict__ qkv,
                                                   float* __restrict__ aout) {
  __shared__ float Ks[KSHORT * 32];
  __shared__ float Vs[KSHORT * 32];
  int d = blockIdx.x >> 2, h = blockIdx.x & 3;
  int t = threadIdx.x;
  for (int e = t; e < KSHORT * 32; e += 128) {
    int tok = e >> 5, c = e & 31;
    const float* p = qkv + ((size_t)(d * KSHORT + tok)) * (3 * RDIM) + h * 32 + c;
    Ks[e] = p[RDIM]; Vs[e] = p[2 * RDIM];
  }
  __syncthreads();
  float q[32];
  const float* qp = qkv + ((size_t)(d * KSHORT + t)) * (3 * RDIM) + h * 32;
#pragma unroll
  for (int c = 0; c < 32; c++) q[c] = qp[c];
  const float scale = 0.17677669529663687f;  // 1/sqrt(32)
  float mx = -3.0e38f;
  for (int k = 0; k < KSHORT; k++) {
    float s = 0.f;
#pragma unroll
    for (int c = 0; c < 32; c++) s += q[c] * Ks[k * 32 + c];
    mx = fmaxf(mx, s * scale);
  }
  float se = 0.f;
  float o[32];
#pragma unroll
  for (int c = 0; c < 32; c++) o[c] = 0.f;
  for (int k = 0; k < KSHORT; k++) {
    float s = 0.f;
#pragma unroll
    for (int c = 0; c < 32; c++) s += q[c] * Ks[k * 32 + c];
    float w = expf(s * scale - mx);
    se += w;
#pragma unroll
    for (int c = 0; c < 32; c++) o[c] += w * Vs[k * 32 + c];
  }
  float inv = 1.f / se;
  float* op = aout + ((size_t)(d * KSHORT + t)) * RDIM + h * 32;
#pragma unroll
  for (int c = 0; c < 32; c++) op[c] = o[c] * inv;
}

__global__ __launch_bounds__(256) void copy_kernel(const float* __restrict__ src,
                                                   float* __restrict__ dst, int n) {
  int i = blockIdx.x * 256 + threadIdx.x;
  if (i < n) dst[i] = src[i];
}

__global__ __launch_bounds__(256) void scatter_kernel(const float* __restrict__ delta,
    const int* __restrict__ idx, const float* __restrict__ rscale, float* __restrict__ out) {
  int t = blockIdx.x * 256 + threadIdx.x;
  if (t >= TOKENS) return;
  int day = t >> 7;
  float rs = fminf(fmaxf(rscale[0], 0.f), 2.f);
  out[(size_t)day * GROUP + idx[t]] += rs * tanhf(delta[t]);  // idx unique per day
}

// ---------------------------------------------------------------------------
// d_in layout (jax pytree leaves, dicts in sorted-key order):
//  0 features | 1 broad_b 2 broad_w 3 gate_b 4 gate_w 5 h1b 6 h1w 7 h2b 8 h2w
//  9 hn_b 10 hn_g 11 in_b 12 in_g 13 lin_scale 14 lin_w
//  15..26 rblocks[0]{ab,ag,b1,b2,fb,fg,inb,inw,ob,ow,w1,w2} 27..38 rblocks[1]
//  39 rin_b 40 rin_w 41 rn_b 42 rn_g 43 rscale
//  44..51 s0.blocks[0]{b1,b2,gb,gw,nb,ng,w1,w2} 52..59 s0.blocks[1]
//  60 s0.tb 61 s0.tbn 62 s0.tg 63 s0.tw
//  64..71 s1.blocks[0] 72..79 s1.blocks[1] 80 s1.tb 81 s1.tbn 82 s1.tg 83 s1.tw
// ---------------------------------------------------------------------------
extern "C" void kernel_launch(void* const* d_in, const int* in_sizes, int n_in,
                              void* d_out, int out_size, void* d_ws, size_t ws_size,
                              hipStream_t stream) {
  (void)in_sizes; (void)n_in; (void)out_size; (void)ws_size;
  auto F = [&](int i) { return (const float*)d_in[i]; };
  char* wsb = (char*)d_ws;
  size_t off = 0;
  auto carve = [&](size_t bytes) -> void* {
    off = (off + 255) & ~(size_t)255;
    void* p = wsb + off;
    off += bytes;
    return p;
  };
  const size_t N = NROWS, T = TOKENS;

  // packed weights (bf16, transposed, K padded)
  u16* gateT = (u16*)carve((size_t)256 * 256 * 2);
  u16* tw0T  = (u16*)carve((size_t)512 * 256 * 2);
  u16* tw1T  = (u16*)carve((size_t)512 * 512 * 2);
  u16 *w1T[4], *w2T[4], *gwT[4];
  for (int i = 0; i < 4; i++) {
    w1T[i] = (u16*)carve((size_t)1024 * 512 * 2);
    w2T[i] = (u16*)carve((size_t)512 * 1024 * 2);
    gwT[i] = (u16*)carve((size_t)512 * 512 * 2);
  }
  u16* rinT = (u16*)carve((size_t)128 * RKPAD * 2);

  // activations
  float* x_f32 = (float*)carve(N * 512 * 4);
  u16*   xb0   = (u16*)carve(N * 512 * 2);
  u16*   xb1   = (u16*)carve(N * 512 * 2);
  u16*   yn    = (u16*)carve(N * 512 * 2);
  u16*   h     = (u16*)carve(N * 1024 * 2);
  float* y     = (float*)carve(N * 512 * 4);
  float* lin   = (float*)carve(N * 4);
  float* broad = (float*)carve(N * 4);
  int*   idxb  = (int*)carve(T * 4);
  u16*   slin  = (u16*)carve(T * RKPAD * 2);
  float* sl    = (float*)carve(T * 128 * 4);
  float* qkv   = (float*)carve(T * 384 * 4);
  float* abuf  = (float*)carve(T * 128 * 4);
  float* aoutb = (float*)carve(T * 128 * 4);
  float* t1    = (float*)carve(T * 256 * 4);
  float* hb    = (float*)carve(T * 128 * 4);
  float* delta = (float*)carve(T * 4);
  // lifetime-safe aliases (save ~256MiB of workspace)
  u16* xn     = (u16*)y;  // [N,256] bf16, dead before y is first written
  u16* x0     = h;        // [N,256] bf16, dead before h is first written
  u16* latent = yn;       // [N,512] bf16, yn dead after last residual block

  float* out = (float*)d_out;
  auto G = [&](size_t n) { return dim3((unsigned)((n + 255) / 256)); };

  // ---- pack weights ----
  pack_w_kernel<<<G((size_t)256 * 256), 256, 0, stream>>>(F(4),  gateT, 256, 256, 256);
  pack_w_kernel<<<G((size_t)512 * 256), 256, 0, stream>>>(F(63), tw0T,  256, 512, 256);
  pack_w_kernel<<<G((size_t)512 * 512), 256, 0, stream>>>(F(83), tw1T,  512, 512, 512);
  const int blkBase[4] = {44, 52, 64, 72};
  for (int i = 0; i < 4; i++) {
    int p = blkBase[i];
    pack_w_kernel<<<G((size_t)1024 * 512), 256, 0, stream>>>(F(p + 6), w1T[i], 512, 1024, 512);
    pack_w_kernel<<<G((size_t)512 * 1024), 256, 0, stream>>>(F(p + 7), w2T[i], 1024, 512, 1024);
    pack_w_kernel<<<G((size_t)512 * 512),  256, 0, stream>>>(F(p + 3), gwT[i], 512, 512, 512);
  }
  pack_w_kernel<<<G((size_t)128 * RKPAD), 256, 0, stream>>>(F(40), rinT, 514, 128, RKPAD);

  // ---- encode + gate + stage transitions + residual blocks (WMMA) ----
  encode_kernel<<<(unsigned)N, 256, 0, stream>>>(F(0), F(12), F(11), F(14), xn, lin);
  // gate GEMM: NOUT=256 -> NB=2 (8 waves x 32 cols = 256, full utilization)
  wmma_mb32_kernel<2, 2><<<(unsigned)(N / 32), 256, 0, stream>>>(
      xn, 256, gateT, F(3), 256, nullptr, x0, xn, nullptr, nullptr);
  // stage-1 transition with fused LN+GELU
  wmma_lngelu_kernel<512><<<(unsigned)(N / 16), 256, 0, stream>>>(
      x0, 256, tw0T, F(60), x_f32, xb0, F(62), F(61));
  u16* cur = xb0;
  for (int bi = 0; bi < 4; bi++) {
    if (bi == 2) {  // stage-2 transition
      u16* tgt = (cur == xb0) ? xb1 : xb0;
      wmma_lngelu_kernel<512><<<(unsigned)(N / 16), 256, 0, stream>>>(
          cur, 512, tw1T, F(80), x_f32, tgt, F(82), F(81));
      cur = tgt;
    }
    int p = blkBase[bi];
    ln_rows_kernel<<<(unsigned)N, 256, 0, stream>>>(x_f32, F(p + 5), F(p + 4), yn, nullptr, 512);
    wmma_mb32_kernel<1, 4><<<(unsigned)(N / 32), 256, 0, stream>>>(
        yn, 512, w1T[bi], F(p + 0), 1024, nullptr, h, nullptr, nullptr, nullptr);
    wmma_mb32_kernel<0, 4><<<(unsigned)(N / 32), 256, 0, stream>>>(
        h, 1024, w2T[bi], F(p + 1), 512, y, nullptr, nullptr, nullptr, nullptr);
    u16* tgt = (cur == xb0) ? xb1 : xb0;
    wmma_mb32_kernel<3, 4><<<(unsigned)(N / 32), 256, 0, stream>>>(
        cur, 512, gwT[bi], F(p + 2), 512, x_f32, tgt, nullptr, x_f32, y);
    cur = tgt;
  }

  // ---- heads + shortlist rerank ----
  latent_broad_kernel<<<(unsigned)N, 256, 0, stream>>>(x_f32, F(10), F(9), F(2), F(1),
                                                       F(13), lin, latent, broad);
  topk_kernel<<<D_DAYS, 256, 0, stream>>>(broad, idxb);
  gather_pack_kernel<<<(unsigned)T, 256, 0, stream>>>(latent, broad, lin, idxb, slin);
  wmma_mb32_kernel<0, 2><<<(unsigned)(T / 32), 256, 0, stream>>>(
      slin, RKPAD, rinT, F(39), 128, sl, nullptr, nullptr, nullptr, nullptr);

  const int rbase[2] = {15, 27};
  for (int r = 0; r < 2; r++) {
    int p = rbase[r];
    ln_rows_kernel<<<(unsigned)T, 256, 0, stream>>>(sl, F(p + 1), F(p + 0), nullptr, abuf, 128);
    gemm_f32_kernel<0><<<G((size_t)T * 384), 256, 0, stream>>>(abuf, F(p + 7), F(p + 6),
                                                               nullptr, qkv, (int)T, 128, 384);
    attn_kernel<<<D_DAYS * 4, 128, 0, stream>>>(qkv, aoutb);
    gemm_f32_kernel<0><<<G((size_t)T * 128), 256, 0, stream>>>(aoutb, F(p + 9), F(p + 8),
                                                               sl, sl, (int)T, 128, 128);
    ln_rows_kernel<<<(unsigned)T, 256, 0, stream>>>(sl, F(p + 5), F(p + 4), nullptr, abuf, 128);
    gemm_f32_kernel<1><<<G((size_t)T * 256), 256, 0, stream>>>(abuf, F(p + 10), F(p + 2),
                                                               nullptr, t1, (int)T, 128, 256);
    gemm_f32_kernel<0><<<G((size_t)T * 128), 256, 0, stream>>>(t1, F(p + 11), F(p + 3),
                                                               sl, sl, (int)T, 256, 128);
  }
  ln_rows_kernel<<<(unsigned)T, 256, 0, stream>>>(sl, F(42), F(41), nullptr, abuf, 128);
  gemm_f32_kernel<1><<<G((size_t)T * 128), 256, 0, stream>>>(abuf, F(6), F(5),
                                                             nullptr, hb, (int)T, 128, 128);
  gemm_f32_kernel<0><<<G((size_t)T), 256, 0, stream>>>(hb, F(8), F(7),
                                                       nullptr, delta, (int)T, 128, 1);
  copy_kernel<<<G(N), 256, 0, stream>>>(broad, out, (int)N);
  scatter_kernel<<<G(T), 256, 0, stream>>>(delta, idxb, F(43), out);
}
